// GCN_35064113004899
// MI455X (gfx1250) — matrix operand order repaired and checked by reading, not compiled, not run on hardware
//
#include <hip/hip_runtime.h>

// ---------------------------------------------------------------------------
// Types for CDNA5 WMMA (wave32): v_wmma_f32_16x16x32_bf16
// ---------------------------------------------------------------------------
typedef __bf16 bf16_t;
typedef __attribute__((ext_vector_type(16))) __bf16 v16bf;
typedef __attribute__((ext_vector_type(8)))  float  v8f;

union FragU { unsigned int u[8]; v16bf v; };   // packed dword view of a bf16 fragment
union FragH { bf16_t h[16];      v16bf v; };   // scalar view of a bf16 fragment

__device__ __forceinline__ v8f wmma_bf16f32(v16bf a, v16bf b, v8f c) {
  // (neg_a, A, neg_b, B, c_mod, C, reuse_a, reuse_b)
  return __builtin_amdgcn_wmma_f32_16x16x32_bf16(false, a, false, b, (short)0, c,
                                                 false, false);
}
__device__ __forceinline__ v8f vzero8() {
  v8f z = {0.f, 0.f, 0.f, 0.f, 0.f, 0.f, 0.f, 0.f};
  return z;
}

// Async global->LDS (gfx1250 ASYNCcnt path), with portable fallback.
#ifndef __has_builtin
#define __has_builtin(x) 0
#endif
#if __has_builtin(__builtin_amdgcn_global_load_async_to_lds_b128)
#define HAVE_ASYNC_LDS 1
typedef __attribute__((ext_vector_type(4))) int v4i_t;   // param type per clang diag
#else
#define HAVE_ASYNC_LDS 0
#endif

// Problem constants
#define NN   2048
#define FF_  4
#define DD   256
#define EE   65536

// ---------------------------------------------------------------------------
// f32 -> bf16 weight/activation conversion
// ---------------------------------------------------------------------------
__global__ void cvt_bf16_kernel(const float* __restrict__ in, bf16_t* __restrict__ out, int n) {
  const int i = blockIdx.x * 256 + threadIdx.x;
  if (i < n) out[i] = (bf16_t)in[i];
}

// ---------------------------------------------------------------------------
// Embedder: x[n,f,d] = features[n,f] * lin_w[d] + lin_b[d]; keep f32 + bf16
// ---------------------------------------------------------------------------
__global__ void embed_kernel(const float* __restrict__ feat, const float* __restrict__ lw,
                             const float* __restrict__ lb, float* __restrict__ x0,
                             bf16_t* __restrict__ xb) {
  const int idx = blockIdx.x * 256 + threadIdx.x;     // N*F*D total
  const int d = idx & 255;
  const int r = idx >> 8;                             // row = n*F + f
  const float v = feat[r] * lw[d] + lb[d];
  x0[idx] = v;
  xb[idx] = (bf16_t)v;
}

// ---------------------------------------------------------------------------
// Fragment loaders (A: 16x32 K-contiguous layout; B: 32x16 K-contiguous layout)
// ---------------------------------------------------------------------------
__device__ __forceinline__ void load_a_frag(FragU& a, const bf16_t* arow, int kb, int akoff) {
  const unsigned int* ap = (const unsigned int*)(arow + kb + akoff);
#pragma unroll
  for (int i = 0; i < 4; ++i) { a.u[i] = ap[i]; a.u[4 + i] = ap[4 + i]; }
}
__device__ __forceinline__ void load_b_frag(FragU& b, const bf16_t* W, size_t col, int K,
                                            int kb, int bkoff) {
  const unsigned int* bp = (const unsigned int*)(W + col * (size_t)K + kb + bkoff);
#pragma unroll
  for (int i = 0; i < 8; ++i) b.u[i] = bp[i];
}

// ---------------------------------------------------------------------------
// Generic WMMA GEMM: C[M,N] = A_bf16[M,K] @ W_bf16[N,K]^T (+bias) (opt relu)
// Block = 4 waves; each wave owns a 16x64 output strip (4 accumulators).
// Branchless software pipeline: always load the next k-step (workspace buffers
// are padded, so the <=128B tail over-read past the logical end is harmless).
// ---------------------------------------------------------------------------
__global__ void __launch_bounds__(128)
wmma_gemm_kernel(const bf16_t* __restrict__ A, const bf16_t* __restrict__ W,
                 const float* __restrict__ bias, float* __restrict__ Cf,
                 bf16_t* __restrict__ Cb, int M, int N, int K, int relu) {
  (void)M;
  const int lane = threadIdx.x & 31;
  const int wave = threadIdx.x >> 5;
  const int l15  = lane & 15;
  const int half = lane >> 4;
  const int tileN   = blockIdx.x * 64;
  const int rowBase = blockIdx.y * 64 + wave * 16;

  v8f acc[4];
#pragma unroll
  for (int t = 0; t < 4; ++t) acc[t] = vzero8();

  const bf16_t* arow = A + (size_t)(rowBase + l15) * K;
  const int akoff = half * 16;   // A fragment: K = akoff+{0..15}
  const int bkoff = half * 16;   // B fragment: K = bkoff+{0..15}

  FragU a0, a1, b0[4], b1[4];
  load_a_frag(a0, arow, 0, akoff);
#pragma unroll
  for (int t = 0; t < 4; ++t)
    load_b_frag(b0[t], W, (size_t)(tileN + t * 16 + l15), K, 0, bkoff);
  load_a_frag(a1, arow, 32, akoff);
#pragma unroll
  for (int t = 0; t < 4; ++t)
    load_b_frag(b1[t], W, (size_t)(tileN + t * 16 + l15), K, 32, bkoff);

  for (int kb = 0; kb < K; kb += 64) {             // K % 64 == 0 for all calls
#pragma unroll
    for (int t = 0; t < 4; ++t) acc[t] = wmma_bf16f32(a0.v, b0[t].v, acc[t]);
    load_a_frag(a0, arow, kb + 64, akoff);         // harmless tail over-read
#pragma unroll
    for (int t = 0; t < 4; ++t)
      load_b_frag(b0[t], W, (size_t)(tileN + t * 16 + l15), K, kb + 64, bkoff);
    __builtin_prefetch(arow + kb + 128, 0, 0);     // global_prefetch_b8
#pragma unroll
    for (int t = 0; t < 4; ++t) acc[t] = wmma_bf16f32(a1.v, b1[t].v, acc[t]);
    load_a_frag(a1, arow, kb + 96, akoff);
#pragma unroll
    for (int t = 0; t < 4; ++t)
      load_b_frag(b1[t], W, (size_t)(tileN + t * 16 + l15), K, kb + 96, bkoff);
  }

#pragma unroll
  for (int t = 0; t < 4; ++t) {
    const int col = tileN + t * 16 + l15;
    const float bv = bias ? bias[col] : 0.f;
#pragma unroll
    for (int j = 0; j < 8; ++j) {
      const int row = rowBase + half * 8 + j;
      float c = acc[t][j] + bv;
      if (relu) c = fmaxf(c, 0.f);
      const size_t o = (size_t)row * N + col;
      if (Cf) Cf[o] = c;
      if (Cb) Cb[o] = (bf16_t)c;
    }
  }
}

// ---------------------------------------------------------------------------
// Fused flash attention. qkv: bf16 [N*F, 768] rows = n*F+f; q|k|v at 0|256|512.
// One wave per (f,h,q-tile of 16); online softmax; 16x32 P tile re-layout via
// LDS; V tiles double-buffered in LDS via async global->LDS copies.
// ---------------------------------------------------------------------------
__device__ __forceinline__ void v_tile_copy(const bf16_t* __restrict__ qkv, bf16_t* ldst,
                                            int lane, int kb, int f, int h) {
  // 32 rows x 64 cols bf16 (128B/row): 8 ops, 4 rows per op, 16B per lane.
#pragma unroll
  for (int i = 0; i < 8; ++i) {
    const int row   = i * 4 + (lane >> 3);
    const int chunk = lane & 7;
    const bf16_t* g =
        qkv + ((size_t)((kb + row) * FF_ + f)) * 768 + 512 + h * 64 + chunk * 8;
    bf16_t* l = ldst + row * 64 + chunk * 8;
#if HAVE_ASYNC_LDS
    __builtin_amdgcn_global_load_async_to_lds_b128((v4i_t*)g, (v4i_t*)l, 0, 0);
#else
    *(uint4*)l = *(const uint4*)g;    // sync fallback (covered by dscnt waits)
#endif
  }
}

__global__ void __launch_bounds__(128)
attn_kernel(const bf16_t* __restrict__ qkv, bf16_t* __restrict__ obf) {
  __shared__ bf16_t Pl[4][16 * 32];
  __shared__ bf16_t Vl[4][2][32 * 64];   // ping-pong V tiles, 8KB per wave
  const int lane = threadIdx.x & 31;
  const int wave = threadIdx.x >> 5;
  const int l15  = lane & 15;
  const int half = lane >> 4;
  const int tile = blockIdx.x * 4 + wave;   // 2048 tiles = 16 (f,h) * 128 q-tiles
  const int fh = tile >> 7;
  const int qt = tile & 127;
  const int f  = fh >> 2;
  const int h  = fh & 3;
  const int qBase = qt * 16;

  // Q fragments (A layout), loop invariant over keys; hd=64 -> 2 k-steps of 32
  FragU qa[2];
  {
    const bf16_t* qp = qkv + ((size_t)((qBase + l15) * FF_ + f)) * 768 + h * 64;
#pragma unroll
    for (int s = 0; s < 2; ++s) {
      const unsigned int* ap = (const unsigned int*)(qp + s * 32 + half * 16);
#pragma unroll
      for (int i = 0; i < 4; ++i) { qa[s].u[i] = ap[i]; qa[s].u[4 + i] = ap[4 + i]; }
    }
  }

  float mrow[8], lrow[8];
  v8f oacc[4];
#pragma unroll
  for (int j = 0; j < 8; ++j) { mrow[j] = -3.0e38f; lrow[j] = 0.f; }
#pragma unroll
  for (int t = 0; t < 4; ++t) oacc[t] = vzero8();

  bf16_t* P = Pl[wave];
  v_tile_copy(qkv, Vl[wave][0], lane, 0, f, h);      // prime first V tile

  for (int kb = 0; kb < NN; kb += 32) {
    const int cur = (kb >> 5) & 1;
    // WAR guard on the ping-pong buffer, then kick off next V tile copy.
    // Always issued (qkv buffer is padded past row NN), keeps the loop branchless.
    asm volatile("s_wait_dscnt 0x0" ::: "memory");
    v_tile_copy(qkv, Vl[wave][cur ^ 1], lane, kb + 32, f, h);

    // ----- S = Q @ K^T for 32 keys (two 16x16 tiles) -----
    v8f s0 = vzero8(), s1 = vzero8();
#pragma unroll
    for (int s = 0; s < 2; ++s) {
      FragU bk0, bk1;
      const unsigned int* kp0 = (const unsigned int*)(
          qkv + ((size_t)((kb + l15) * FF_ + f)) * 768 + 256 + h * 64 + s * 32 + half * 16);
      const unsigned int* kp1 = (const unsigned int*)(
          qkv + ((size_t)((kb + 16 + l15) * FF_ + f)) * 768 + 256 + h * 64 + s * 32 + half * 16);
#pragma unroll
      for (int i = 0; i < 8; ++i) { bk0.u[i] = kp0[i]; bk1.u[i] = kp1[i]; }
      s0 = wmma_bf16f32(qa[s].v, bk0.v, s0);
      s1 = wmma_bf16f32(qa[s].v, bk1.v, s1);
    }

    // ----- online softmax: row stats replicated across each 16-lane half -----
    float p0[8], p1[8];
#pragma unroll
    for (int j = 0; j < 8; ++j) {
      const float a = s0[j] * 0.125f;   // 1/sqrt(64)
      const float b = s1[j] * 0.125f;
      float mx = fmaxf(a, b);
#pragma unroll
      for (int m = 1; m < 16; m <<= 1) mx = fmaxf(mx, __shfl_xor(mx, m, 16));
      const float mnew = fmaxf(mrow[j], mx);
      const float e0 = __expf(a - mnew);
      const float e1 = __expf(b - mnew);
      float rs = e0 + e1;
#pragma unroll
      for (int m = 1; m < 16; m <<= 1) rs += __shfl_xor(rs, m, 16);
      const float corr = __expf(mrow[j] - mnew);
      lrow[j] = lrow[j] * corr + rs;
      mrow[j] = mnew;
#pragma unroll
      for (int t = 0; t < 4; ++t) oacc[t][j] *= corr;
      p0[j] = e0;
      p1[j] = e1;
    }

    // ----- re-layout P (C layout -> A fragment) through LDS -----
#pragma unroll
    for (int j = 0; j < 8; ++j) {
      const int r = half * 8 + j;
      P[r * 32 + l15]      = (bf16_t)p0[j];
      P[r * 32 + 16 + l15] = (bf16_t)p1[j];
    }
    asm volatile("s_wait_dscnt 0x0" ::: "memory");  // intra-wave LDS handoff
    FragU pa;
    {
      const unsigned int* pu = (const unsigned int*)(P + l15 * 32 + half * 16);
#pragma unroll
      for (int i = 0; i < 4; ++i) { pa.u[i] = pu[i]; pa.u[4 + i] = pu[4 + i]; }
    }

    // ----- wait for current V tile; async completes in-order, so <=8
    // outstanding means everything older (the current tile) has landed. -----
#if HAVE_ASYNC_LDS
#if __has_builtin(__builtin_amdgcn_s_wait_asynccnt)
    __builtin_amdgcn_s_wait_asynccnt(8);
#else
    asm volatile("s_wait_asynccnt 0x8" ::: "memory");
#endif
#else
    asm volatile("s_wait_dscnt 0x0" ::: "memory");  // fallback used ds stores
#endif

    // ----- O += P @ V (four 16-wide column tiles over hd=64, V from LDS) -----
    const bf16_t* Vc = Vl[wave][cur];
#pragma unroll
    for (int t = 0; t < 4; ++t) {
      FragH bv;
      const int coln = t * 16 + l15;
#pragma unroll
      for (int i = 0; i < 16; ++i)
        bv.h[i] = Vc[(half * 16 + i) * 64 + coln];
      oacc[t] = wmma_bf16f32(pa.v, bv.v, oacc[t]);
    }
  }

  // ----- finalize: O /= l, write bf16 [N*F, 256] for out-proj GEMM -----
#pragma unroll
  for (int j = 0; j < 8; ++j) {
    const float inv = 1.f / lrow[j];
    const int q = qBase + half * 8 + j;
#pragma unroll
    for (int t = 0; t < 4; ++t) {
      obf[((size_t)(q * FF_ + f)) * 256 + h * 64 + t * 16 + l15] =
          (bf16_t)(oacc[t][j] * inv);
    }
  }
}

// ---------------------------------------------------------------------------
// Residual add + LayerNorm over D=256 (one row per block)
// ---------------------------------------------------------------------------
__global__ void __launch_bounds__(256)
addln_kernel(const float* __restrict__ xin, const float* __restrict__ res,
             const float* __restrict__ g, const float* __restrict__ b,
             float* __restrict__ xo, bf16_t* __restrict__ xob) {
  __shared__ float red[256];
  const int r = blockIdx.x, t = threadIdx.x;
  const size_t base = (size_t)r * 256;
  const float v = xin[base + t] + res[base + t];
  red[t] = v;
  __syncthreads();
  for (int s = 128; s > 0; s >>= 1) { if (t < s) red[t] += red[t + s]; __syncthreads(); }
  const float mean = red[0] * (1.f / 256.f);
  __syncthreads();
  const float dv = v - mean;
  red[t] = dv * dv;
  __syncthreads();
  for (int s = 128; s > 0; s >>= 1) { if (t < s) red[t] += red[t + s]; __syncthreads(); }
  const float rstd = rsqrtf(red[0] * (1.f / 256.f) + 1e-5f);
  const float o = dv * rstd * g[t] + b[t];
  if (xo)  xo[base + t] = o;
  if (xob) xob[base + t] = (bf16_t)o;
}

// Mean over feature-batch dim + relu -> pooled bf16 [N, 256]
__global__ void __launch_bounds__(256)
pool_kernel(const float* __restrict__ x2, bf16_t* __restrict__ pooled) {
  const int n = blockIdx.x, d = threadIdx.x;
  float s = 0.f;
#pragma unroll
  for (int f = 0; f < FF_; ++f) s += x2[((size_t)(n * FF_ + f)) * 256 + d];
  pooled[(size_t)n * 256 + d] = (bf16_t)fmaxf(s * 0.25f, 0.f);
}

// ---------------------------------------------------------------------------
// GCN support
// ---------------------------------------------------------------------------
__global__ void deg_init_kernel(float* deg) { deg[blockIdx.x * 256 + threadIdx.x] = 1.f; }

__global__ void deg_count_kernel(const int* __restrict__ edges, float* deg) {
  const int e = blockIdx.x * 256 + threadIdx.x;   // E = 65536
  atomicAdd(&deg[edges[EE + e]], 1.f);
}

__global__ void dinv_kernel(float* deg) {
  const int i = blockIdx.x * 256 + threadIdx.x;
  deg[i] = rsqrtf(deg[i]);                        // deg >= 1 (self loops)
}

// out[n,c] = bias[c] + h[n,c] * dinv[n]^2   (self-loop term)
__global__ void gcn_init_kernel(const float* __restrict__ h, const float* __restrict__ dinv,
                                const float* __restrict__ bias, float* __restrict__ out) {
  const int idx = blockIdx.x * 256 + threadIdx.x;
  const int n = idx >> 9, c = idx & 511;
  const float w = dinv[n];
  out[idx] = bias[c] + h[idx] * w * w;
}

// out[dst] += h[src] * dinv[src]*dinv[dst]; one edge per block, 512 cols
__global__ void __launch_bounds__(256)
gcn_scatter_kernel(const int* __restrict__ edges, const float* __restrict__ h,
                   const float* __restrict__ dinv, float* __restrict__ out) {
  const int e = blockIdx.x;
  const int s = edges[e];
  const int d = edges[EE + e];
  const float nm = dinv[s] * dinv[d];
  const int c = threadIdx.x;
  atomicAdd(&out[(size_t)d * 512 + c],       h[(size_t)s * 512 + c] * nm);
  atomicAdd(&out[(size_t)d * 512 + c + 256], h[(size_t)s * 512 + c + 256] * nm);
}

__global__ void gcn_relu_bf16_kernel(const float* __restrict__ in, bf16_t* __restrict__ outb) {
  const int idx = blockIdx.x * 256 + threadIdx.x;
  outb[idx] = (bf16_t)fmaxf(in[idx], 0.f);
}

// ---------------------------------------------------------------------------
// Host orchestration
// ---------------------------------------------------------------------------
extern "C" void kernel_launch(void* const* d_in, const int* in_sizes, int n_in,
                              void* d_out, int out_size, void* d_ws, size_t ws_size,
                              hipStream_t stream) {
  (void)in_sizes; (void)n_in; (void)out_size; (void)ws_size;
  const float* feat = (const float*)d_in[0];
  const int*   edges = (const int*)d_in[1];
  const float* lin_w = (const float*)d_in[2];
  const float* lin_b = (const float*)d_in[3];
  const float* ipw  = (const float*)d_in[4];
  const float* ipb  = (const float*)d_in[5];
  const float* outw = (const float*)d_in[6];
  const float* outb = (const float*)d_in[7];
  const float* ln1g = (const float*)d_in[8];
  const float* ln1b = (const float*)d_in[9];
  const float* fw1  = (const float*)d_in[10];
  const float* fb1  = (const float*)d_in[11];
  const float* fw2  = (const float*)d_in[12];
  const float* fb2  = (const float*)d_in[13];
  const float* ln2g = (const float*)d_in[14];
  const float* ln2b = (const float*)d_in[15];
  const float* W1 = (const float*)d_in[16];
  const float* b1 = (const float*)d_in[17];
  const float* W2 = (const float*)d_in[18];
  const float* b2 = (const float*)d_in[19];
  const float* W3 = (const float*)d_in[20];
  const float* b3 = (const float*)d_in[21];
  const float* W4 = (const float*)d_in[22];
  const float* b4 = (const float*)d_in[23];
  float* dout = (float*)d_out;

  char* wsp = (char*)d_ws;
  // Every buffer gets 1KB of slack so branchless pipelines may over-read tails.
  auto alloc = [&](size_t bytes) -> void* {
    void* p = (void*)wsp;
    wsp += (bytes + 1024 + 255) & ~(size_t)255;
    return p;
  };

  const size_t R = (size_t)NN * FF_;   // 8192 rows
  float*  x0    = (float*)alloc(R * DD * 4);
  bf16_t* xb    = (bf16_t*)alloc(R * DD * 2);
  bf16_t* wqkv  = (bf16_t*)alloc((size_t)768 * 256 * 2);
  bf16_t* wout  = (bf16_t*)alloc((size_t)256 * 256 * 2);
  bf16_t* wf1   = (bf16_t*)alloc((size_t)2048 * 256 * 2);
  bf16_t* wf2   = (bf16_t*)alloc((size_t)256 * 2048 * 2);
  bf16_t* wg1   = (bf16_t*)alloc((size_t)512 * 256 * 2);
  bf16_t* wg2   = (bf16_t*)alloc((size_t)512 * 512 * 2);
  bf16_t* wg3   = (bf16_t*)alloc((size_t)512 * 512 * 2);
  bf16_t* wg4   = (bf16_t*)alloc((size_t)512 * 512 * 2);
  // qkvb padded by one 32-key tile (32*FF_*768 elems) for the always-issued
  // next-tile async copy in the attention loop.
  bf16_t* qkvb  = (bf16_t*)alloc(R * 768 * 2 + (size_t)32 * FF_ * 768 * 2);
  bf16_t* ob    = (bf16_t*)alloc(R * 256 * 2);
  float*  oproj = (float*)alloc(R * 256 * 4);
  float*  x1    = (float*)alloc(R * 256 * 4);
  bf16_t* x1b   = (bf16_t*)alloc(R * 256 * 2);
  bf16_t* ffh   = (bf16_t*)alloc(R * 2048 * 2);
  float*  ff2   = (float*)alloc(R * 256 * 4);
  float*  x2    = (float*)alloc(R * 256 * 4);
  bf16_t* pooled= (bf16_t*)alloc((size_t)NN * 256 * 2);
  float*  deg   = (float*)alloc((size_t)NN * 4);
  float*  hbuf  = (float*)alloc((size_t)NN * 512 * 4);
  float*  agg   = (float*)alloc((size_t)NN * 512 * 4);
  bf16_t* xga   = (bf16_t*)alloc((size_t)NN * 512 * 2);
  bf16_t* xgb   = (bf16_t*)alloc((size_t)NN * 512 * 2);

  auto cvt = [&](const float* src, bf16_t* dst, int n) {
    cvt_bf16_kernel<<<(n + 255) / 256, 256, 0, stream>>>(src, dst, n);
  };
  auto gemm = [&](const bf16_t* A, const bf16_t* Wm, const float* bias,
                  float* Cf, bf16_t* Cb, int M, int N, int K, int relu) {
    dim3 g(N / 64, M / 64);
    wmma_gemm_kernel<<<g, 128, 0, stream>>>(A, Wm, bias, Cf, Cb, M, N, K, relu);
  };

  // ---- weight conversion (once per launch) ----
  cvt(ipw,  wqkv, 768 * 256);
  cvt(outw, wout, 256 * 256);
  cvt(fw1,  wf1,  2048 * 256);
  cvt(fw2,  wf2,  256 * 2048);
  cvt(W1,   wg1,  512 * 256);
  cvt(W2,   wg2,  512 * 512);
  cvt(W3,   wg3,  512 * 512);
  cvt(W4,   wg4,  512 * 512);

  // ---- embed ----
  embed_kernel<<<(int)(R * DD / 256), 256, 0, stream>>>(feat, lin_w, lin_b, x0, xb);

  // ---- transformer encoder layer ----
  gemm(xb, wqkv, ipb, nullptr, qkvb, (int)R, 768, 256, 0);           // QKV
  attn_kernel<<<512, 128, 0, stream>>>(qkvb, ob);                    // flash attention
  gemm(ob, wout, outb, oproj, nullptr, (int)R, 256, 256, 0);         // out-proj
  addln_kernel<<<(int)R, 256, 0, stream>>>(oproj, x0, ln1g, ln1b, x1, x1b);
  gemm(x1b, wf1, fb1, nullptr, ffh, (int)R, 2048, 256, 1);           // FFN1 + relu
  gemm(ffh, wf2, fb2, ff2, nullptr, (int)R, 256, 2048, 0);           // FFN2
  addln_kernel<<<(int)R, 256, 0, stream>>>(ff2, x1, ln2g, ln2b, x2, nullptr);
  pool_kernel<<<NN, 256, 0, stream>>>(x2, pooled);                   // mean+relu

  // ---- GCN degree normalization ----
  deg_init_kernel<<<NN / 256, 256, 0, stream>>>(deg);
  deg_count_kernel<<<EE / 256, 256, 0, stream>>>(edges, deg);
  dinv_kernel<<<NN / 256, 256, 0, stream>>>(deg);

  // ---- 4 GCNConv layers ----
  auto gcn = [&](const bf16_t* X, int K, const bf16_t* Wg, const float* bias,
                 float* acc, bf16_t* nextb) {
    gemm(X, Wg, nullptr, hbuf, nullptr, NN, 512, K, 0);
    gcn_init_kernel<<<NN * 512 / 256, 256, 0, stream>>>(hbuf, deg, bias, acc);
    gcn_scatter_kernel<<<EE, 256, 0, stream>>>(edges, hbuf, deg, acc);
    if (nextb)
      gcn_relu_bf16_kernel<<<NN * 512 / 256, 256, 0, stream>>>(acc, nextb);
  };
  gcn(pooled, 256, wg1, b1, agg, xga);
  gcn(xga,    512, wg2, b2, agg, xgb);
  gcn(xgb,    512, wg3, b3, agg, xga);
  gcn(xga,    512, wg4, b4, dout, nullptr);   // final layer: no relu, direct to d_out
}
